// HybridDiTBlock_70145405878732
// MI455X (gfx1250) — matrix operand order repaired and checked by reading, not compiled, not run on hardware
//
#include <hip/hip_runtime.h>
#include <hip/hip_bf16.h>
#include <math.h>

// ---------------------------------------------------------------------------
// Problem constants
// ---------------------------------------------------------------------------
#define BSZ   2
#define LSEQ  2048
#define DIMC  1024
#define DINC  2048      // EXP * DIM
#define DSTC  16
#define DTRC  64        // DIM/16
#define FFC   4096
#define SIXD  (6 * DIMC)
#define TWO_DIN (2 * DINC)

// CDNA5 vector types for WMMA
typedef __attribute__((ext_vector_type(16))) __bf16 bf16x16;
typedef __attribute__((ext_vector_type(8)))  __bf16 bf16x8;
typedef __attribute__((ext_vector_type(8)))  float  f32x8;

// ---------------------------------------------------------------------------
// Workspace layout (floats, all offsets multiples of 1024 for alignment)
// ---------------------------------------------------------------------------
#define EMB_OFS   ((size_t)0)                         // 2*6144
#define NORM_OFS  ((size_t)16384)                     // 4096*1024 (reused for norm2)
#define XC_OFS    (NORM_OFS + (size_t)4194304)        // 4096*2048
#define ZS_OFS    (XC_OFS   + (size_t)8388608)        // 4096*2048
#define DBC_OFS   (ZS_OFS   + (size_t)8388608)        // 4096*96
#define DT_OFS    (DBC_OFS  + (size_t)393216)         // 4096*2048
#define YSS_OFS   (DT_OFS   + (size_t)8388608)        // 4096*2048
#define X2_OFS    (YSS_OFS  + (size_t)8388608)        // 4096*1024
#define XZ_OFS    (X2_OFS   + (size_t)4194304)        // 4096*4096 (reused for h1)

__device__ __forceinline__ float silu_f(float v) {
    return v / (1.0f + __expf(-v));
}

// ---------------------------------------------------------------------------
// K1: emb = silu(t) @ W_ada + b_ada     (B x 6144)
// ---------------------------------------------------------------------------
__global__ __launch_bounds__(256) void ada_kernel(
    const float* __restrict__ t, const float* __restrict__ W,
    const float* __restrict__ bias, float* __restrict__ emb)
{
    __shared__ float st[DIMC];
    const int b = blockIdx.y;
    const int o = blockIdx.x * 256 + threadIdx.x;
    for (int i = threadIdx.x; i < DIMC; i += 256)
        st[i] = silu_f(t[(size_t)b * DIMC + i]);
    __syncthreads();
    float acc = bias[o];
    for (int k = 0; k < DIMC; ++k)
        acc += st[k] * W[(size_t)k * SIXD + o];
    emb[(size_t)b * SIXD + o] = acc;
}

// ---------------------------------------------------------------------------
// K2/K9: norm = LN(x) * (1 + scale) + shift   (one block per row)
// ---------------------------------------------------------------------------
__global__ __launch_bounds__(256) void ln_mod_kernel(
    const float* __restrict__ x, const float* __restrict__ emb,
    int shiftOfs, int scaleOfs, float* __restrict__ out)
{
    const int row = blockIdx.x;
    const int b = row / LSEQ;
    const float* xr = x + (size_t)row * DIMC;
    __shared__ float rbuf[16];
    float vals[4];
    float s = 0.f, s2 = 0.f;
#pragma unroll
    for (int i = 0; i < 4; ++i) {
        float v = xr[threadIdx.x + i * 256];
        vals[i] = v; s += v; s2 += v * v;
    }
#pragma unroll
    for (int o = 16; o > 0; o >>= 1) {
        s  += __shfl_down(s,  o, 32);
        s2 += __shfl_down(s2, o, 32);
    }
    if ((threadIdx.x & 31) == 0) {
        rbuf[threadIdx.x >> 5]     = s;
        rbuf[8 + (threadIdx.x >> 5)] = s2;
    }
    __syncthreads();
    float S = 0.f, S2 = 0.f;
#pragma unroll
    for (int w = 0; w < 8; ++w) { S += rbuf[w]; S2 += rbuf[8 + w]; }
    const float mean = S * (1.0f / DIMC);
    const float var  = S2 * (1.0f / DIMC) - mean * mean;
    const float rstd = rsqrtf(var + 1e-6f);
    const float* er = emb + (size_t)b * SIXD;
#pragma unroll
    for (int i = 0; i < 4; ++i) {
        int col = threadIdx.x + i * 256;
        float sc = er[scaleOfs + col];
        float sh = er[shiftOfs + col];
        out[(size_t)row * DIMC + col] = (vals[i] - mean) * rstd * (1.0f + sc) + sh;
    }
}

// ---------------------------------------------------------------------------
// WMMA tiled GEMM: C(MxN) = epilogue(A(MxK) @ B(KxN) [+ bias])
//   bf16 on-the-fly conversion, f32 accumulation via v_wmma_f32_16x16x32_bf16
//   Block tile 128x128, BK=32, 8 waves (wave32), each wave 32x64.
// EPI: 0=store, 1=tanh-gelu, 2=softplus, 3=res + gate*(v)
// ---------------------------------------------------------------------------
#define TILE_M 128
#define TILE_N 128
#define TILE_K 32
#define LDSS   48     // padded LDS stride (bf16 units) -> 96B rows, 16B aligned

template<int EPI, bool BIAS>
__global__ __launch_bounds__(256) void wmma_gemm_kernel(
    const float* __restrict__ A, int lda,
    const float* __restrict__ Bg, int ldb,
    const float* __restrict__ bias,
    float* __restrict__ C, int ldc,
    const float* __restrict__ res,
    const float* __restrict__ gate, int gstride,
    int M, int N, int K, int lseq)
{
    __shared__ __bf16 As[TILE_M * LDSS];
    __shared__ __bf16 Bs[TILE_N * LDSS];

    const int tid  = threadIdx.x;
    const int lane = tid & 31;
    const int wid  = tid >> 5;       // 0..7
    const int wm   = wid & 3;        // 4 waves along M
    const int wn   = wid >> 2;       // 2 waves along N
    const int m0   = blockIdx.y * TILE_M;
    const int n0   = blockIdx.x * TILE_N;
    const int wm0  = wm * 32;
    const int wn0  = wn * 64;
    const int lm   = lane & 15;
    const int lh   = lane >> 4;

    f32x8 acc[2][4] = {};

    const bool nfull = (n0 + TILE_N) <= N;
    const int arow = tid >> 1;
    const int acg  = (tid & 1) * 16;
    const int bkr  = tid >> 3;
    const int bnc  = (tid & 7) * 16;

    for (int k0 = 0; k0 < K; k0 += TILE_K) {
        // ---- stage A tile (f32 -> bf16), row-major [128][48]
        {
            const float* aSrc = A + (size_t)(m0 + arow) * lda + (k0 + acg);
            __bf16* aDst = As + arow * LDSS + acg;
#pragma unroll
            for (int i = 0; i < 4; ++i) {
                float4 v = *(const float4*)(aSrc + i * 4);
                aDst[i*4+0] = (__bf16)v.x; aDst[i*4+1] = (__bf16)v.y;
                aDst[i*4+2] = (__bf16)v.z; aDst[i*4+3] = (__bf16)v.w;
            }
        }
        // ---- stage B tile transposed (f32 -> bf16), [N=128][K=32] in [128][48]
        {
            const float* bSrc = Bg + (size_t)(k0 + bkr) * ldb + n0 + bnc;
            if (nfull) {
#pragma unroll
                for (int i = 0; i < 4; ++i) {
                    float4 v = *(const float4*)(bSrc + i * 4);
                    Bs[(bnc + i*4 + 0) * LDSS + bkr] = (__bf16)v.x;
                    Bs[(bnc + i*4 + 1) * LDSS + bkr] = (__bf16)v.y;
                    Bs[(bnc + i*4 + 2) * LDSS + bkr] = (__bf16)v.z;
                    Bs[(bnc + i*4 + 3) * LDSS + bkr] = (__bf16)v.w;
                }
            } else {
#pragma unroll
                for (int i = 0; i < 16; ++i) {
                    int n = n0 + bnc + i;
                    float v = (n < N) ? bSrc[i] : 0.0f;
                    Bs[(bnc + i) * LDSS + bkr] = (__bf16)v;
                }
            }
        }
        __syncthreads();

        // prefetch next K tile (global_prefetch_b8)
        if (k0 + TILE_K < K) {
            __builtin_prefetch(A + (size_t)(m0 + arow) * lda + (k0 + TILE_K + acg), 0, 0);
            __builtin_prefetch(Bg + (size_t)(k0 + TILE_K + bkr) * ldb + n0 + bnc, 0, 0);
        }

        // ---- A fragments: lane=row, half-lane selects K 0-15 / 16-31 (contiguous)
        bf16x16 af[2];
#pragma unroll
        for (int mi = 0; mi < 2; ++mi) {
            const __bf16* p = As + (wm0 + mi * 16 + lm) * LDSS + lh * 16;
            bf16x8 lo = *(const bf16x8*)p;
            bf16x8 hi = *(const bf16x8*)(p + 8);
            bf16x16 a;
#pragma unroll
            for (int e = 0; e < 8; ++e) { a[e] = lo[e]; a[e + 8] = hi[e]; }
            af[mi] = a;
        }
        // ---- B fragments: lane=column, half-lane selects K 0-15 / 16-31 (contiguous)
#pragma unroll
        for (int ni = 0; ni < 4; ++ni) {
            const __bf16* p = Bs + (wn0 + ni * 16 + lm) * LDSS + lh * 16;
            bf16x16 bfr = *(const bf16x16*)p;
#pragma unroll
            for (int mi = 0; mi < 2; ++mi) {
                acc[mi][ni] = __builtin_amdgcn_wmma_f32_16x16x32_bf16(
                    false, af[mi], false, bfr, (short)0, acc[mi][ni], false, false);
            }
        }
        __syncthreads();
    }

    // ---- epilogue: VGPR r, lanes 0-15 -> M=r, lanes 16-31 -> M=8+r; N=lane&15
    const int bidx = m0 / lseq;
#pragma unroll
    for (int mi = 0; mi < 2; ++mi) {
#pragma unroll
        for (int ni = 0; ni < 4; ++ni) {
            const int Ng = n0 + wn0 + ni * 16 + lm;
            if (Ng >= N) continue;
            const int Mb = m0 + wm0 + mi * 16 + lh * 8;
            const float bv = BIAS ? bias[Ng] : 0.0f;
            const float gv = (EPI == 3) ? gate[(size_t)bidx * gstride + Ng] : 0.0f;
#pragma unroll
            for (int r = 0; r < 8; ++r) {
                float v = acc[mi][ni][r] + bv;
                const size_t off = (size_t)(Mb + r) * ldc + Ng;
                if (EPI == 1) {
                    float u = v;
                    v = 0.5f * u * (1.0f + tanhf(0.79788456080286535588f *
                                                  (u + 0.044715f * u * u * u)));
                } else if (EPI == 2) {
                    v = (v > 20.0f) ? v : log1pf(__expf(v));
                } else if (EPI == 3) {
                    v = res[off] + gv * v;
                }
                C[off] = v;
            }
        }
    }
}

// ---------------------------------------------------------------------------
// K4: causal depthwise conv (DCONV=4) + silu; also zs = silu(z)
// ---------------------------------------------------------------------------
__global__ __launch_bounds__(256) void conv_silu_kernel(
    const float* __restrict__ xz, const float* __restrict__ cw,
    const float* __restrict__ cb, float* __restrict__ xc,
    float* __restrict__ zs)
{
    const size_t idx = (size_t)blockIdx.x * 256 + threadIdx.x;     // B*L*DIN
    const int d = (int)(idx & (DINC - 1));
    const size_t row = idx >> 11;
    const int l = (int)(row & (LSEQ - 1));
    float acc = cb[d];
#pragma unroll
    for (int k = 0; k < 4; ++k) {
        int ls = l - (3 - k);
        if (ls >= 0)
            acc += cw[d * 4 + k] * xz[(row - (size_t)(3 - k)) * TWO_DIN + d];
    }
    xc[idx] = silu_f(acc);
    float z = xz[row * TWO_DIN + DINC + d];
    zs[idx] = silu_f(z);
}

// ---------------------------------------------------------------------------
// K7: selective-state scan, one thread per (b, d) channel, 16-wide state
//     ySS = (y + D*xc) * silu(z)
// ---------------------------------------------------------------------------
__global__ __launch_bounds__(256) void scan_kernel(
    const float* __restrict__ dt, const float* __restrict__ xc,
    const float* __restrict__ dbc, const float* __restrict__ A_log,
    const float* __restrict__ Dv, const float* __restrict__ zs,
    float* __restrict__ ySS)
{
    const int d = blockIdx.x * 256 + threadIdx.x;
    const int b = blockIdx.y;
    float Aj[DSTC], h[DSTC];
#pragma unroll
    for (int j = 0; j < DSTC; ++j) {
        Aj[j] = -__expf(A_log[(size_t)d * DSTC + j]);
        h[j] = 0.0f;
    }
    const float Dd = Dv[d];
    const size_t rbase = (size_t)b * LSEQ;
    for (int l = 0; l < LSEQ; ++l) {
        const size_t row = rbase + l;
        const float dtv = dt[row * DINC + d];
        const float xv  = xc[row * DINC + d];
        const float dx  = dtv * xv;
        const float* bc = dbc + row * 96;   // [.. :64]=dt_raw, [64:80]=B, [80:96]=C
        float y = 0.0f;
#pragma unroll
        for (int j = 0; j < DSTC; ++j) {
            h[j] = __expf(dtv * Aj[j]) * h[j] + dx * bc[64 + j];
            y += h[j] * bc[80 + j];
        }
        ySS[row * DINC + d] = (y + Dd * xv) * zs[row * DINC + d];
    }
}

// ---------------------------------------------------------------------------
// launch
// ---------------------------------------------------------------------------
extern "C" void kernel_launch(void* const* d_in, const int* in_sizes, int n_in,
                              void* d_out, int out_size, void* d_ws, size_t ws_size,
                              hipStream_t stream) {
    const float* x      = (const float*)d_in[0];
    const float* t      = (const float*)d_in[1];
    const float* W_ada  = (const float*)d_in[2];
    const float* b_ada  = (const float*)d_in[3];
    const float* W_in   = (const float*)d_in[4];
    const float* conv_w = (const float*)d_in[5];
    const float* conv_b = (const float*)d_in[6];
    const float* W_xprj = (const float*)d_in[7];
    const float* W_dt   = (const float*)d_in[8];
    const float* b_dt   = (const float*)d_in[9];
    const float* A_log  = (const float*)d_in[10];
    const float* Dv     = (const float*)d_in[11];
    const float* W_out  = (const float*)d_in[12];
    const float* W1     = (const float*)d_in[13];
    const float* b1     = (const float*)d_in[14];
    const float* W2     = (const float*)d_in[15];
    const float* b2     = (const float*)d_in[16];
    float* out = (float*)d_out;
    float* ws  = (float*)d_ws;

    float* emb  = ws + EMB_OFS;
    float* norm = ws + NORM_OFS;   // reused as norm2
    float* xcb  = ws + XC_OFS;
    float* zsb  = ws + ZS_OFS;
    float* dbc  = ws + DBC_OFS;
    float* dtb  = ws + DT_OFS;
    float* ySS  = ws + YSS_OFS;
    float* x2   = ws + X2_OFS;
    float* xz   = ws + XZ_OFS;     // reused as h1

    const int M = BSZ * LSEQ;      // 4096

    // K1: adaLN embedding
    ada_kernel<<<dim3(SIXD / 256, BSZ), 256, 0, stream>>>(t, W_ada, b_ada, emb);

    // K2: LN + (scale_msa, shift_msa)   chunks: shift=0, scale=1
    ln_mod_kernel<<<dim3(M), 256, 0, stream>>>(x, emb, 0 * DIMC, 1 * DIMC, norm);

    // K3: xz = norm @ W_in    (4096 x 4096, K=1024)
    wmma_gemm_kernel<0, false><<<dim3(TWO_DIN / TILE_N, M / TILE_M), 256, 0, stream>>>(
        norm, DIMC, W_in, TWO_DIN, nullptr, xz, TWO_DIN,
        nullptr, nullptr, 0, M, TWO_DIN, DIMC, LSEQ);

    // K4: causal conv + silu, zs = silu(z)
    conv_silu_kernel<<<dim3((M * DINC) / 256), 256, 0, stream>>>(
        xz, conv_w, conv_b, xcb, zsb);

    // K5: dbc = xc @ W_xproj   (4096 x 96, K=2048)
    wmma_gemm_kernel<0, false><<<dim3(1, M / TILE_M), 256, 0, stream>>>(
        xcb, DINC, W_xprj, 96, nullptr, dbc, 96,
        nullptr, nullptr, 0, M, 96, DINC, LSEQ);

    // K6: dt = softplus(dbc[:, :64] @ W_dt + b_dt)   (4096 x 2048, K=64)
    wmma_gemm_kernel<2, true><<<dim3(DINC / TILE_N, M / TILE_M), 256, 0, stream>>>(
        dbc, 96, W_dt, DINC, b_dt, dtb, DINC,
        nullptr, nullptr, 0, M, DINC, DTRC, LSEQ);

    // K7: SSM scan
    scan_kernel<<<dim3(DINC / 256, BSZ), 256, 0, stream>>>(
        dtb, xcb, dbc, A_log, Dv, zsb, ySS);

    // K8: x2 = x + gate_msa * (ySS @ W_out)   (4096 x 1024, K=2048)
    wmma_gemm_kernel<3, false><<<dim3(DIMC / TILE_N, M / TILE_M), 256, 0, stream>>>(
        ySS, DINC, W_out, DIMC, nullptr, x2, DIMC,
        x, emb + 2 * DIMC, SIXD, M, DIMC, DINC, LSEQ);

    // K9: norm2 = LN(x2) with (scale_mlp, shift_mlp)  chunks: shift=3, scale=4
    ln_mod_kernel<<<dim3(M), 256, 0, stream>>>(x2, emb, 3 * DIMC, 4 * DIMC, norm);

    // K10: h1 = gelu_tanh(norm2 @ W1 + b1)   (4096 x 4096, K=1024)
    wmma_gemm_kernel<1, true><<<dim3(FFC / TILE_N, M / TILE_M), 256, 0, stream>>>(
        norm, DIMC, W1, FFC, b1, xz, FFC,
        nullptr, nullptr, 0, M, FFC, DIMC, LSEQ);

    // K11: out = x2 + gate_mlp * (h1 @ W2 + b2)   (4096 x 1024, K=4096)
    wmma_gemm_kernel<3, true><<<dim3(DIMC / TILE_N, M / TILE_M), 256, 0, stream>>>(
        xz, FFC, W2, DIMC, b2, out, DIMC,
        x2, emb + 5 * DIMC, SIXD, M, DIMC, FFC, LSEQ);
}
